// Contextual_Attention_11690900980457
// MI455X (gfx1250) — matrix operand order repaired and checked by reading, not compiled, not run on hardware
//
#include <hip/hip_runtime.h>
#include <math.h>

typedef __attribute__((ext_vector_type(16))) __bf16 v16bf;
typedef __attribute__((ext_vector_type(8)))  float  v8f;
typedef unsigned short u16;
typedef long long ll;

__device__ __forceinline__ u16 f2bf(float f) {
  unsigned u = __float_as_uint(f);
  unsigned r = u + 0x7FFFu + ((u >> 16) & 1u);   // round-to-nearest-even
  return (u16)(r >> 16);
}

union BF16Frag { v16bf v; u16 u[16]; };

// ---------------------------------------------------------------------------
// bf16 WMMA GEMM:  C = A * B^T (+bias) (+Add)
//   A : bf16 [M x Kdim] row-major (lda), or implicit im2col of feacat (CONVK>0)
//   Bt: bf16 [N x Kdim] row-major (ldb)  -- K-major panels, contiguous frags
//   Epilogue: optional bias[N], fp32 Add, fp32 store, bf16 store
//             (bf16 store transposed per batch when TRANSC).
// 8 waves/block; each wave computes a 32x64 tile (2x4 WMMA tiles,
// 8 wmma / k-step from 2 A-frags + 4 B-frags = 1.5 b128 loads per wmma).
// Block tile: 64 (M) x 256 (N).
// ---------------------------------------------------------------------------
template<int CONVK, bool TRANSC>
__global__ __launch_bounds__(256) void gemm_wmma(
    const u16* __restrict__ A, ll lda, ll strideA,
    const u16* __restrict__ Bt, ll ldb, ll strideB,
    const float* __restrict__ bias,
    u16* __restrict__ Cb, ll ldcb, ll strideCb,
    float* __restrict__ Cf, ll ldcf, ll strideCf,
    const float* __restrict__ Add,
    int M, int N, int Kdim, int Sdim)
{
  const int lane = threadIdx.x & 31;
  const int wave = threadIdx.x >> 5;
  const int half = lane >> 4;     // lane-half (wave32)
  const int l16  = lane & 15;
  const int tile_m = blockIdx.y * 64  + (wave & 1) * 32;
  const int tile_n = blockIdx.x * 256 + (wave >> 1) * 64;
  if (tile_m >= M || tile_n >= N) return;   // wave-uniform guard
  const int bz = blockIdx.z;
  A  += (ll)bz * strideA;
  Bt += (ll)bz * strideB;

  int arow[2], cb[2], cs[2], bcol[4];
  #pragma unroll
  for (int t = 0; t < 2; ++t) {
    arow[t] = tile_m + t * 16 + l16;
    if (CONVK > 0) { cb[t] = arow[t] / Sdim; cs[t] = arow[t] - cb[t] * Sdim; }
  }
  #pragma unroll
  for (int t = 0; t < 4; ++t) bcol[t] = tile_n + t * 16 + l16;

  v8f acc[2][4];
  #pragma unroll
  for (int i = 0; i < 2; ++i)
    #pragma unroll
    for (int j = 0; j < 4; ++j)
      #pragma unroll
      for (int e = 0; e < 8; ++e) acc[i][j][e] = 0.0f;

  for (int k0 = 0; k0 < Kdim; k0 += 32) {
    BF16Frag af[2];
    // ---- A fragments: 16x32 bf16; two contiguous 8-elem chunks per lane ----
    #pragma unroll
    for (int t = 0; t < 2; ++t) {
      if (CONVK == 0) {
        const u16* ap = A + (ll)arow[t] * lda + k0 + half * 8;
        #pragma unroll
        for (int j = 0; j < 8; ++j) af[t].u[j]     = ap[j];
        #pragma unroll
        for (int j = 0; j < 8; ++j) af[t].u[8 + j] = ap[16 + j];
      } else {
        #pragma unroll
        for (int c = 0; c < 2; ++c) {
          int kb   = k0 + c * 16 + half * 8;   // 8-aligned; never crosses a tap
          int tap  = kb >> 10;
          int ch   = kb & 1023;
          int srow = cs[t] + tap - (CONVK / 2);   // SAME zero padding
          const u16* ap = A + ((ll)(cb[t] * Sdim + srow)) * lda + ch;
          if (srow >= 0 && srow < Sdim) {
            #pragma unroll
            for (int j = 0; j < 8; ++j) af[t].u[c * 8 + j] = ap[j];
          } else {
            #pragma unroll
            for (int j = 0; j < 8; ++j) af[t].u[c * 8 + j] = (u16)0;
          }
        }
      }
    }
    // ---- B fragments: 32x16 bf16, two contiguous 8-elem K chunks/lane; consume
    //      each fragment immediately with its two WMMAs ----
    #pragma unroll
    for (int j = 0; j < 4; ++j) {
      BF16Frag bf;
      const u16* bp = Bt + (ll)bcol[j] * ldb + k0 + half * 8;
      #pragma unroll
      for (int e = 0; e < 16; ++e) bf.u[e] = bp[e < 8 ? e : e + 8];
      #pragma unroll
      for (int i = 0; i < 2; ++i)
        acc[i][j] = __builtin_amdgcn_wmma_f32_16x16x32_bf16(
            false, af[i].v, false, bf.v, (short)0, acc[i][j], false, false);
    }
  }

  // ---- epilogue: C/D layout M = 8*half + e, N = l16 ----
  #pragma unroll
  for (int i = 0; i < 2; ++i) {
    #pragma unroll
    for (int j = 0; j < 4; ++j) {
      #pragma unroll
      for (int e = 0; e < 8; ++e) {
        int rowg = tile_m + i * 16 + half * 8 + e;
        int colg = tile_n + j * 16 + l16;
        float v = acc[i][j][e];
        if (bias) v += bias[colg];
        if (Add)  v += Add[(ll)bz * strideCf + (ll)rowg * ldcf + colg];
        if (Cf)   Cf[(ll)bz * strideCf + (ll)rowg * ldcf + colg] = v;
        if (Cb) {
          if (TRANSC) {  // store [b][colg][s] (per-batch transposed)
            int b2 = rowg / Sdim, s2 = rowg - (rowg / Sdim) * Sdim;
            Cb[((ll)b2 * ldcb + colg) * (ll)Sdim + s2] = f2bf(v);
          } else {
            Cb[(ll)bz * strideCb + (ll)rowg * ldcb + colg] = f2bf(v);
          }
        }
      }
    }
  }
}

// ---------------------------------------------------------------------------
// Masked softmax, one wave per score row of length S=1024 (32 elems/lane).
// ---------------------------------------------------------------------------
__global__ __launch_bounds__(256) void softmax_rows(
    const float* __restrict__ scores, u16* __restrict__ P,
    const int* __restrict__ seqlen, float scale, int S)
{
  const int lane = threadIdx.x & 31;
  const int wave = threadIdx.x >> 5;
  const int row  = blockIdx.x * 8 + wave;
  const int b    = blockIdx.y;
  const int len  = seqlen[b];
  const float* sr = scores + ((ll)b * S + row) * (ll)S;
  u16*         pr = P      + ((ll)b * S + row) * (ll)S;

  float v[32];
  float m = -INFINITY;
  #pragma unroll
  for (int i = 0; i < 32; ++i) {
    int k = i * 32 + lane;
    float x = sr[k] * scale;
    v[i] = (k < len) ? x : -INFINITY;
    m = fmaxf(m, v[i]);
  }
  #pragma unroll
  for (int off = 16; off > 0; off >>= 1)
    m = fmaxf(m, __shfl_xor(m, off, 32));

  float sum = 0.0f;
  #pragma unroll
  for (int i = 0; i < 32; ++i) {
    float e = (v[i] == -INFINITY) ? 0.0f : __expf(v[i] - m);
    v[i] = e;
    sum += e;
  }
  #pragma unroll
  for (int off = 16; off > 0; off >>= 1)
    sum += __shfl_xor(sum, off, 32);

  float inv = 1.0f / sum;   // len >= 1 guaranteed -> sum > 0
  #pragma unroll
  for (int i = 0; i < 32; ++i)
    pr[i * 32 + lane] = f2bf(v[i] * inv);
}

// ---------------------------------------------------------------------------
// fp32 -> bf16 cast with output row stride
// ---------------------------------------------------------------------------
__global__ void cast_strided(const float* __restrict__ x, u16* __restrict__ y,
                             ll n, int cols, int ldout)
{
  ll idx = (ll)blockIdx.x * blockDim.x + threadIdx.x;
  if (idx >= n) return;
  ll r = idx / cols;
  int c = (int)(idx - r * cols);
  y[r * (ll)ldout + c] = f2bf(x[idx]);
}

// cn_w[O=1024][I=1024][taps] fp32 -> w2[o][t*1024+i] bf16 (K-major im2col panel)
__global__ void conv_w_prep(const float* __restrict__ w, u16* __restrict__ w2,
                            int taps)
{
  ll idx = (ll)blockIdx.x * blockDim.x + threadIdx.x;
  ll total = (ll)1024 * 1024 * taps;
  if (idx >= total) return;
  int o   = (int)(idx / (1024 * taps));
  int rem = (int)(idx - (ll)o * (1024 * taps));
  int i   = rem / taps;
  int t   = rem - i * taps;
  w2[(ll)o * (1024 * taps) + (ll)t * 1024 + i] = f2bf(w[idx]);
}

// ---------------------------------------------------------------------------
extern "C" void kernel_launch(void* const* d_in, const int* in_sizes, int n_in,
                              void* d_out, int out_size, void* d_ws, size_t ws_size,
                              hipStream_t stream)
{
  (void)in_sizes; (void)n_in; (void)out_size; (void)ws_size;
  const int B = 16, S = 1024, D = 1024;
  const ll BS = (ll)B * S;

  const float* feaQK  = (const float*)d_in[0];
  const float* feaV   = (const float*)d_in[1];
  const int*   seqlen = (const int*)  d_in[2];
  const float* cn3_w  = (const float*)d_in[3];
  const float* cn3_b  = (const float*)d_in[4];
  const float* cn5_w  = (const float*)d_in[5];
  const float* cn5_b  = (const float*)d_in[6];
  const float* k_w    = (const float*)d_in[7];
  const float* k_b    = (const float*)d_in[8];
  const float* q_w    = (const float*)d_in[9];
  const float* q_b    = (const float*)d_in[10];
  const float* v_w    = (const float*)d_in[11];
  const float* v_b    = (const float*)d_in[12];

  char* ws = (char*)d_ws;
  auto alloc = [&](size_t bytes) -> char* {
    char* p = ws; ws += (bytes + 255) & ~(size_t)255; return p;
  };
  u16*   feacat = (u16*)  alloc(BS * 3072 * 2);            // [feaQK | k3 | k5] bf16
  u16*   feaVb  = (u16*)  alloc(BS * D * 2);
  u16*   Qb     = (u16*)  alloc(BS * D * 2);
  u16*   Kb     = (u16*)  alloc(BS * D * 2);
  u16*   Vt     = (u16*)  alloc(BS * D * 2);               // V^T per batch [b][d][s]
  float* Vf     = (float*)alloc(BS * D * 4);               // V fp32, [b*s][d]
  float* Sc     = (float*)alloc((ll)B * S * S * 4);        // raw scores fp32
  u16*   Pb     = (u16*)  alloc((ll)B * S * S * 2);        // probabilities bf16
  u16*   qwb    = (u16*)  alloc((ll)D * D * 2);            // q_w bf16 [out][in]
  u16*   kwb    = (u16*)  alloc((ll)D * 3072 * 2);         // k_w bf16 [out][in]
  u16*   vwb    = (u16*)  alloc((ll)D * D * 2);
  u16*   c3w2   = (u16*)  alloc((ll)3 * D * D * 2);        // [o][t*1024+i]
  u16*   c5w2   = (u16*)  alloc((ll)5 * D * D * 2);

  const ll nQK = BS * D;
  // --- casts & weight prep (weights stay [N][K] = K-major panels) ---
  cast_strided<<<dim3((unsigned)((nQK + 255) / 256)), 256, 0, stream>>>(feaQK, feacat, nQK, D, 3072);
  cast_strided<<<dim3((unsigned)((nQK + 255) / 256)), 256, 0, stream>>>(feaV,  feaVb,  nQK, D, D);
  cast_strided<<<dim3((D * D + 255) / 256),    256, 0, stream>>>(q_w, qwb, (ll)D * D, D, D);
  cast_strided<<<dim3((D * D + 255) / 256),    256, 0, stream>>>(v_w, vwb, (ll)D * D, D, D);
  cast_strided<<<dim3((D * 3072 + 255) / 256), 256, 0, stream>>>(k_w, kwb, (ll)D * 3072, 3072, 3072);
  conv_w_prep <<<dim3((3 * D * D + 255) / 256), 256, 0, stream>>>(cn3_w, c3w2, 3);
  conv_w_prep <<<dim3((5 * D * D + 255) / 256), 256, 0, stream>>>(cn5_w, c5w2, 5);

  dim3 gProj(4, 256, 1);    // N/256, (B*S)/64
  dim3 gAttn(4, 16, 16);    // S/256, S/64, B

  // Q = feaQK @ q_w^T + q_b                       (bf16 out)
  gemm_wmma<0, false><<<gProj, 256, 0, stream>>>(
      feacat, 3072, 0, qwb, 1024, 0, q_b,
      Qb, 1024, 0, nullptr, 0, 0, nullptr, 16384, 1024, 1024, S);
  // k3 = conv1d_k3(feaQK) -> feacat cols [1024,2048)
  gemm_wmma<3, false><<<gProj, 256, 0, stream>>>(
      feacat, 3072, 0, c3w2, 3072, 0, cn3_b,
      feacat + 1024, 3072, 0, nullptr, 0, 0, nullptr, 16384, 1024, 3072, S);
  // k5 = conv1d_k5(feaQK) -> feacat cols [2048,3072)
  gemm_wmma<5, false><<<gProj, 256, 0, stream>>>(
      feacat, 3072, 0, c5w2, 5120, 0, cn5_b,
      feacat + 2048, 3072, 0, nullptr, 0, 0, nullptr, 16384, 1024, 5120, S);
  // K = feacat @ k_w^T + k_b                      (bf16 out)
  gemm_wmma<0, false><<<gProj, 256, 0, stream>>>(
      feacat, 3072, 0, kwb, 3072, 0, k_b,
      Kb, 1024, 0, nullptr, 0, 0, nullptr, 16384, 1024, 3072, S);
  // V = feaV @ v_w^T + v_b     (bf16 out TRANSPOSED per batch + fp32 out)
  gemm_wmma<0, true><<<gProj, 256, 0, stream>>>(
      feaVb, 1024, 0, vwb, 1024, 0, v_b,
      Vt, 1024 /*=D*/, 0, Vf, 1024, 0, nullptr, 16384, 1024, 1024, S);
  // scores = Q @ K^T  (per batch, fp32 out, unscaled)
  gemm_wmma<0, false><<<gAttn, 256, 0, stream>>>(
      Qb, 1024, (ll)S * D, Kb, 1024, (ll)S * D, nullptr,
      nullptr, 0, 0, Sc, 1024, (ll)S * S, nullptr, 1024, 1024, 1024, S);
  // atten = softmax(scale * scores, key mask)  -> bf16 P
  softmax_rows<<<dim3(S / 8, B), 256, 0, stream>>>(Sc, Pb, seqlen, 0.03125f, S);
  // out = atten @ V + V   (fp32 to d_out);  B^T = Vt [b][d][s]
  gemm_wmma<0, false><<<gAttn, 256, 0, stream>>>(
      Pb, 1024, (ll)S * S, Vt, 1024, (ll)D * S, nullptr,
      nullptr, 0, 0, (float*)d_out, 1024, (ll)S * D, Vf,
      1024, 1024, 1024, S);
}
